// ResidualMambaLayer_77206332113657
// MI455X (gfx1250) — matrix operand order repaired and checked by reading, not compiled, not run on hardware
//
#include <hip/hip_runtime.h>
#include <stdint.h>

// ---------------------------------------------------------------------------
// Mamba residual layer for MI455X (gfx1250, wave32, WMMA bf16 + TDM)
// ---------------------------------------------------------------------------

#define BATCH   4
#define SEQLEN  2048
#define D_MODEL 1024
#define D_INNER 2048
#define DT_RANK 64
#define D_STATE 16
#define NX      96          // DT_RANK + 2*D_STATE
#define NXP     128         // NX padded to a multiple of 64 (GEMM tile width)
#define BL      (BATCH * SEQLEN)   // 8192 rows

typedef __attribute__((ext_vector_type(16))) __bf16        v16bf;
typedef __attribute__((ext_vector_type(8)))  float         v8f;
typedef __attribute__((ext_vector_type(8)))  unsigned int  v8u;
typedef __attribute__((ext_vector_type(4)))  unsigned int  v4u;
typedef __attribute__((ext_vector_type(8)))  int           v8i;
typedef __attribute__((ext_vector_type(4)))  int           v4i;

__device__ __forceinline__ unsigned short f2bfbits(float f) {
    unsigned u = __builtin_bit_cast(unsigned, f);
    unsigned r = u + 0x7FFFu + ((u >> 16) & 1u);   // round-to-nearest-even
    return (unsigned short)(r >> 16);
}

__device__ __forceinline__ float silu(float v) {
    return v / (1.0f + __expf(-v));
}

// ---------------------------------------------------------------------------
// TDM: 2D-tile DMA (bf16 elements) global -> LDS.
// Tile: tile_d0 = 32 elements (one k-tile) x tile_d1 = 128 rows,
// row stride = strideElems. Descriptor per CDNA5 ISA.
// This toolchain's builtin takes 6 args: (g0, g1, g2, g3, g4, cpol).
// ---------------------------------------------------------------------------
__device__ __forceinline__ void tdm_load_a_slab(uint32_t lds_addr,
                                                const unsigned short* gptr,
                                                uint32_t strideElems)
{
    const uint64_t ga = (uint64_t)(uintptr_t)gptr;
    v4u g0;
    g0[0] = 1u;                                          // count=1, user mode
    g0[1] = lds_addr;                                    // lds_addr [63:32]
    g0[2] = (uint32_t)(ga & 0xFFFFFFFFu);                // global_addr[31:0]
    g0[3] = (uint32_t)((ga >> 32) & 0x01FFFFFFu)         // global_addr[56:32]
          | (2u << 30);                                  // type = 2 ("image")
    const uint32_t td0 = 1u << 30;                       // tensor_dim0 (OOB: huge)
    const uint32_t td1 = 1u << 20;                       // tensor_dim1
    v8i g1;
    g1[0] = (int)(1u << 16);                             // data_size=1 -> 2 bytes
    g1[1] = (int)((td0 & 0xFFFFu) << 16);                // tensor_dim0[15:0] @63:48
    g1[2] = (int)(((td0 >> 16) & 0xFFFFu)                // tensor_dim0[31:16]
          | ((td1 & 0xFFFFu) << 16));                    // tensor_dim1[15:0]
    g1[3] = (int)(((td1 >> 16) & 0xFFFFu)                // tensor_dim1[31:16]
          | (32u << 16));                                // tile_dim0 = 32 @127:112
    g1[4] = (int)128u;                                   // tile_dim1 = 128; tile_dim2=0
    g1[5] = (int)strideElems;                            // tensor_dim0_stride[31:0]
    g1[6] = 0;                                           // stride[47:32]=0, dim1_stride lo=0
    g1[7] = 0;
    const v4i z4 = {0, 0, 0, 0};
    const v8i z8 = {0, 0, 0, 0, 0, 0, 0, 0};
    __builtin_amdgcn_tensor_load_to_lds(g0, g1, z4, z4, z8, 0);
}

// ---------------------------------------------------------------------------
// Pack a row-major f32 weight (K x N) into WMMA-B-fragment-native bf16 tiles,
// padded to Npad columns (zero fill). Tile = 32(K) x 16(N); lane-major so each
// lane's 16 bf16 are contiguous (2x b128 loads). Layout [nTile][kTile].
// ---------------------------------------------------------------------------
__global__ __launch_bounds__(256)
void pack_weights_kernel(const float* __restrict__ W, unsigned short* __restrict__ P,
                         int K, int N, int Npad)
{
    const int kTiles = K >> 5;
    const size_t total = (size_t)K * Npad;
    for (size_t p = (size_t)blockIdx.x * blockDim.x + threadIdx.x;
         p < total; p += (size_t)gridDim.x * blockDim.x) {
        size_t tile  = p >> 9;
        int within   = (int)(p & 511);
        int lane     = within >> 4;
        int e        = within & 15;
        int nT       = (int)(tile / kTiles);
        int kT       = (int)(tile % kTiles);
        int n        = nT * 16 + (lane & 15);
        int k        = kT * 32 + (lane >> 4) * 16 + e;
        float v      = (n < N) ? W[(size_t)k * N + n] : 0.0f;
        P[p] = f2bfbits(v);
    }
}

// ---------------------------------------------------------------------------
// LayerNorm over D_MODEL, output bf16 row-major (WMMA A operand)
// ---------------------------------------------------------------------------
__global__ __launch_bounds__(128)
void layernorm_kernel(const float* __restrict__ x,
                      const float* __restrict__ w,
                      const float* __restrict__ b,
                      unsigned short* __restrict__ out)
{
    const int row = blockIdx.x;
    const float* xr = x + (size_t)row * D_MODEL;
    float s = 0.f, s2 = 0.f;
    for (int i = threadIdx.x; i < D_MODEL; i += 128) {
        float v = xr[i]; s += v; s2 += v * v;
    }
    #pragma unroll
    for (int o = 16; o > 0; o >>= 1) { s += __shfl_down(s, o); s2 += __shfl_down(s2, o); }
    __shared__ float red[2][4];
    const int wv = threadIdx.x >> 5, ln = threadIdx.x & 31;
    if (ln == 0) { red[0][wv] = s; red[1][wv] = s2; }
    __syncthreads();
    s  = red[0][0] + red[0][1] + red[0][2] + red[0][3];
    s2 = red[1][0] + red[1][1] + red[1][2] + red[1][3];
    const float mu   = s * (1.0f / D_MODEL);
    const float var  = s2 * (1.0f / D_MODEL) - mu * mu;
    const float rstd = rsqrtf(var + 1e-5f);
    for (int i = threadIdx.x; i < D_MODEL; i += 128) {
        float v = (xr[i] - mu) * rstd * w[i] + b[i];
        out[(size_t)row * D_MODEL + i] = f2bfbits(v);
    }
}

// ---------------------------------------------------------------------------
// WMMA bf16 GEMM: C(MxN f32) = A(MxK bf16 row-major) x Bp(packed bf16 tiles)
// Block: 256 thr = 8 waves; block tile 128(M) x 64(N); wave tile 16 x 64.
// A slab (128x32 bf16, 8KB) DMA'd by the Tensor Data Mover, double-buffered:
// TDM of k-tile t+1 overlaps WMMA on k-tile t. No bounds checks in the hot
// loop (N is padded to a multiple of 64 at pack time).
// ---------------------------------------------------------------------------
__global__ __launch_bounds__(256)
void gemm_bf16_wmma_kernel(const unsigned short* __restrict__ A,
                           const unsigned short* __restrict__ Bp,
                           float* __restrict__ C,
                           const float* __restrict__ addend,
                           int M, int N, int K)
{
    __shared__ unsigned short As[2][128 * 32];   // 2 x 8 KB slabs
    const int tid    = threadIdx.x;
    const int wave   = tid >> 5;
    const int lane   = tid & 31;
    const int r      = lane & 15;
    const int half   = lane >> 4;
    const int mBase  = blockIdx.x * 128;
    const int nBase  = blockIdx.y * 64;
    const int kTiles = K >> 5;
    const int ntBase = nBase >> 4;

    v8f acc[4];
    #pragma unroll
    for (int i = 0; i < 4; ++i)
        #pragma unroll
        for (int v = 0; v < 8; ++v) acc[i][v] = 0.0f;

    const unsigned short* Abase = A + (size_t)mBase * K;

    // prologue: DMA k-tile 0 into slab 0
    if (wave == 0)
        tdm_load_a_slab((uint32_t)(uintptr_t)&As[0][0], Abase, (uint32_t)K);

    for (int kt = 0; kt < kTiles; ++kt) {
        const int buf = kt & 1;
        if (wave == 0) {
            if (kt + 1 < kTiles) {
                tdm_load_a_slab((uint32_t)(uintptr_t)&As[buf ^ 1][0],
                                Abase + (size_t)(kt + 1) * 32, (uint32_t)K);
                __builtin_amdgcn_s_wait_tensorcnt(1);   // current slab ready
            } else {
                __builtin_amdgcn_s_wait_tensorcnt(0);
            }
        }
        __syncthreads();

        // A fragment (contiguous k = 16*half + s, matching packed B): two adjacent b128 runs
        const uint4* ald = (const uint4*)(As[buf] + (wave * 16 + r) * 32);
        const uint4 a0 = ald[2 * half];
        const uint4 a1 = ald[2 * half + 1];
        v8u au; au[0]=a0.x; au[1]=a0.y; au[2]=a0.z; au[3]=a0.w;
                au[4]=a1.x; au[5]=a1.y; au[6]=a1.z; au[7]=a1.w;
        const v16bf afrag = __builtin_bit_cast(v16bf, au);

        #pragma unroll
        for (int i = 0; i < 4; ++i) {
            const uint4* bsrc = (const uint4*)(Bp + ((size_t)(ntBase + i) * kTiles + kt) * 512
                                                  + lane * 16);
            const uint4 b0 = bsrc[0];
            const uint4 b1 = bsrc[1];
            v8u bu; bu[0]=b0.x; bu[1]=b0.y; bu[2]=b0.z; bu[3]=b0.w;
                    bu[4]=b1.x; bu[5]=b1.y; bu[6]=b1.z; bu[7]=b1.w;
            const v16bf bfrag = __builtin_bit_cast(v16bf, bu);
            acc[i] = __builtin_amdgcn_wmma_f32_16x16x32_bf16(
                        false, afrag, false, bfrag, (short)0, acc[i], false, false);
        }
        __syncthreads();
    }

    // store: VGPR v, lanes 0-15 -> M=v, lanes 16-31 -> M=v+8 ; N = lane&15
    const int mRow = mBase + wave * 16 + 8 * half;
    if (addend) {
        #pragma unroll
        for (int i = 0; i < 4; ++i) {
            const int n = nBase + i * 16 + r;
            #pragma unroll
            for (int v = 0; v < 8; ++v) {
                const size_t off = (size_t)(mRow + v) * N + n;
                C[off] = acc[i][v] + addend[off];
            }
        }
    } else {
        #pragma unroll
        for (int i = 0; i < 4; ++i) {
            const int n = nBase + i * 16 + r;
            #pragma unroll
            for (int v = 0; v < 8; ++v) {
                const size_t off = (size_t)(mRow + v) * N + n;
                C[off] = acc[i][v];
            }
        }
    }
}

// ---------------------------------------------------------------------------
// Depthwise causal conv (D_CONV=4) + bias + SiLU over u = xz[:, :D_INNER].
// ---------------------------------------------------------------------------
__global__ __launch_bounds__(256)
void conv_silu_kernel(const float* __restrict__ xz,
                      const float* __restrict__ cw,
                      const float* __restrict__ cb,
                      float* __restrict__ uf,
                      unsigned short* __restrict__ ub)
{
    const size_t idx = (size_t)blockIdx.x * blockDim.x + threadIdx.x;
    if (idx >= (size_t)BL * D_INNER) return;
    const int d      = (int)(idx % D_INNER);
    const size_t row = idx / D_INNER;
    const int l      = (int)(row % SEQLEN);
    const float* base = xz + row * (2 * D_INNER) + d;
    float acc = cb[d];
    #pragma unroll
    for (int i = 0; i < 4; ++i) {
        const int li = l - 3 + i;
        if (li >= 0) acc += cw[d * 4 + i] * base[(ptrdiff_t)(li - l) * (2 * D_INNER)];
    }
    const float sv = silu(acc);
    uf[row * D_INNER + d] = sv;
    ub[row * D_INNER + d] = f2bfbits(sv);
}

// dt_in = dbl[:, :DT_RANK] -> bf16 (A operand of dt projection GEMM)
__global__ __launch_bounds__(256)
void dtin_bf16_kernel(const float* __restrict__ dbl, unsigned short* __restrict__ out)
{
    const size_t idx = (size_t)blockIdx.x * blockDim.x + threadIdx.x;
    if (idx >= (size_t)BL * DT_RANK) return;
    const size_t row = idx / DT_RANK;
    const int c = (int)(idx % DT_RANK);
    out[idx] = f2bfbits(dbl[row * NXP + c]);
}

// dt = softplus(dtraw + b_dt), in place
__global__ __launch_bounds__(256)
void softplus_kernel(float* __restrict__ dt, const float* __restrict__ b_dt)
{
    const size_t idx = (size_t)blockIdx.x * blockDim.x + threadIdx.x;
    if (idx >= (size_t)BL * D_INNER) return;
    const int d = (int)(idx % D_INNER);
    const float v = dt[idx] + b_dt[d];
    dt[idx] = (v > 20.0f) ? v : log1pf(__expf(v));
}

// ---------------------------------------------------------------------------
// Selective scan: one thread per (batch, channel), 16 states in registers;
// B_t/C_t staged through LDS in 64-step chunks. Fuses +u*D and *silu(z),
// emits bf16 for the output GEMM.
// ---------------------------------------------------------------------------
__global__ __launch_bounds__(256)
void scan_kernel(const float* __restrict__ dbl,   // (BL, NXP): cols 64..79=B, 80..95=C
                 const float* __restrict__ dt,
                 const float* __restrict__ u,
                 const float* __restrict__ xz,
                 const float* __restrict__ A_log,
                 const float* __restrict__ Dp,
                 unsigned short* __restrict__ yb)
{
    const int d = blockIdx.x * 256 + threadIdx.x;
    const int b = blockIdx.y;
    __shared__ float Bs[64][D_STATE];
    __shared__ float Cs[64][D_STATE];

    float Av[D_STATE];
    #pragma unroll
    for (int n = 0; n < D_STATE; ++n) Av[n] = -__expf(A_log[(size_t)d * D_STATE + n]);
    const float Dv = Dp[d];

    float h[D_STATE];
    #pragma unroll
    for (int n = 0; n < D_STATE; ++n) h[n] = 0.0f;

    for (int c = 0; c < SEQLEN; c += 64) {
        __syncthreads();
        for (int i = threadIdx.x; i < 64 * 32; i += 256) {
            const int s = i >> 5, w = i & 31;
            const float v = dbl[((size_t)b * SEQLEN + c + s) * NXP + DT_RANK + w];
            if (w < D_STATE) Bs[s][w] = v; else Cs[s][w - D_STATE] = v;
        }
        __syncthreads();

        for (int s = 0; s < 64; ++s) {
            const size_t row = (size_t)b * SEQLEN + c + s;
            const float dtv = dt[row * D_INNER + d];
            const float uv  = u [row * D_INNER + d];
            const float du  = dtv * uv;
            float y = 0.0f;
            #pragma unroll
            for (int n = 0; n < D_STATE; ++n) {
                const float dA = __expf(dtv * Av[n]);
                h[n] = dA * h[n] + du * Bs[s][n];
                y += h[n] * Cs[s][n];
            }
            const float zv = xz[row * (2 * D_INNER) + D_INNER + d];
            const float yf = (y + uv * Dv) * silu(zv);
            yb[row * D_INNER + d] = f2bfbits(yf);
        }
    }
}

// ---------------------------------------------------------------------------
extern "C" void kernel_launch(void* const* d_in, const int* in_sizes, int n_in,
                              void* d_out, int out_size, void* d_ws, size_t ws_size,
                              hipStream_t stream)
{
    const float* x      = (const float*)d_in[0];
    const float* ln_w   = (const float*)d_in[1];
    const float* ln_b   = (const float*)d_in[2];
    const float* W_in   = (const float*)d_in[3];
    const float* conv_w = (const float*)d_in[4];
    const float* conv_b = (const float*)d_in[5];
    const float* W_x    = (const float*)d_in[6];
    const float* W_dt   = (const float*)d_in[7];
    const float* b_dt   = (const float*)d_in[8];
    const float* A_log  = (const float*)d_in[9];
    const float* Dvec   = (const float*)d_in[10];
    const float* W_out  = (const float*)d_in[11];
    float* out = (float*)d_out;

    char* ws = (char*)d_ws;
    size_t off = 0;
    auto alloc = [&](size_t bytes) -> void* {
        void* p = ws + off;
        off = (off + bytes + 255) & ~(size_t)255;
        return p;
    };

    unsigned short* WinP  = (unsigned short*)alloc((size_t)D_MODEL * 2 * D_INNER * 2);
    unsigned short* WxP   = (unsigned short*)alloc((size_t)D_INNER * NXP * 2);
    unsigned short* WdtP  = (unsigned short*)alloc((size_t)DT_RANK * D_INNER * 2);
    unsigned short* WoutP = (unsigned short*)alloc((size_t)D_INNER * D_MODEL * 2);
    unsigned short* xn_bf = (unsigned short*)alloc((size_t)BL * D_MODEL * 2);
    float*          xz    = (float*)alloc((size_t)BL * 2 * D_INNER * 4);
    float*          u_f   = (float*)alloc((size_t)BL * D_INNER * 4);
    unsigned short* u_bf  = (unsigned short*)alloc((size_t)BL * D_INNER * 2);
    float*          dbl   = (float*)alloc((size_t)BL * NXP * 4);
    unsigned short* dtinb = (unsigned short*)alloc((size_t)BL * DT_RANK * 2);
    float*          dt    = (float*)alloc((size_t)BL * D_INNER * 4);
    unsigned short* y_bf  = (unsigned short*)alloc((size_t)BL * D_INNER * 2);

    // 1) pack weights to WMMA-native bf16 tiles (W_x padded 96 -> 128 cols)
    pack_weights_kernel<<<4096, 256, 0, stream>>>(W_in,  WinP,  D_MODEL, 2 * D_INNER, 2 * D_INNER);
    pack_weights_kernel<<<1024, 256, 0, stream>>>(W_x,   WxP,   D_INNER, NX,          NXP);
    pack_weights_kernel<<<1024, 256, 0, stream>>>(W_dt,  WdtP,  DT_RANK, D_INNER,     D_INNER);
    pack_weights_kernel<<<4096, 256, 0, stream>>>(W_out, WoutP, D_INNER, D_MODEL,     D_MODEL);

    // 2) layernorm -> bf16
    layernorm_kernel<<<BL, 128, 0, stream>>>(x, ln_w, ln_b, xn_bf);

    // 3) xz = xn @ W_in          (8192 x 4096, K=1024)
    gemm_bf16_wmma_kernel<<<dim3(BL / 128, (2 * D_INNER) / 64), 256, 0, stream>>>(
        xn_bf, WinP, xz, nullptr, BL, 2 * D_INNER, D_MODEL);

    // 4) depthwise causal conv + SiLU
    conv_silu_kernel<<<(BL * D_INNER) / 256, 256, 0, stream>>>(xz, conv_w, conv_b, u_f, u_bf);

    // 5) dbl = u @ W_x           (8192 x 128(padded 96), K=2048)
    gemm_bf16_wmma_kernel<<<dim3(BL / 128, NXP / 64), 256, 0, stream>>>(
        u_bf, WxP, dbl, nullptr, BL, NXP, D_INNER);

    // 6) dt = softplus(dt_in @ W_dt + b_dt)   (8192 x 2048, K=64)
    dtin_bf16_kernel<<<(BL * DT_RANK) / 256, 256, 0, stream>>>(dbl, dtinb);
    gemm_bf16_wmma_kernel<<<dim3(BL / 128, D_INNER / 64), 256, 0, stream>>>(
        dtinb, WdtP, dt, nullptr, BL, D_INNER, DT_RANK);
    softplus_kernel<<<(BL * D_INNER) / 256, 256, 0, stream>>>(dt, b_dt);

    // 7) selective scan (+ u*D, * silu(z) gating) -> bf16
    scan_kernel<<<dim3(D_INNER / 256, BATCH), 256, 0, stream>>>(
        dbl, dt, u_f, xz, A_log, Dvec, y_bf);

    // 8) out = x + y @ W_out     (8192 x 1024, K=2048), residual fused
    gemm_bf16_wmma_kernel<<<dim3(BL / 128, D_MODEL / 64), 256, 0, stream>>>(
        y_bf, WoutP, out, x, BL, D_MODEL, D_INNER);

    (void)in_sizes; (void)n_in; (void)out_size; (void)ws_size;
}
